// FastKNNCell_80247168958985
// MI455X (gfx1250) — matrix operand order repaired and checked by reading, not compiled, not run on hardware
//
#include <hip/hip_runtime.h>
#include <hip/hip_bf16.h>

// SRU-like fused cell for gfx1250 (MI455X).
//   u = x @ W   (M=32 batch, N=3072 = 1024 outputs x 3 gates, K=1024)
//   c_t = f*c_{t-1} + (1-f)*u0 ; h = r*tanh(c) + (1-r)*x   (scan over L=1024)
// grid = 128 blocks = (2 batch-tiles) x (64 output-tiles); block = 4 waves
// K-splitting the 1024-deep reduction (256 each -> 8-deep WMMA chains,
// 512 waves total). The block's 48 weight columns live in LDS for all 1024
// timesteps (staged once via the Tensor Data Mover). Partials combine through
// a double-buffered LDS reduction with one barrier per timestep; recurrent
// state c stays in the consumer wave's registers.

typedef __attribute__((ext_vector_type(16))) __bf16 v16bf;
typedef __attribute__((ext_vector_type(8)))  float  v8f;
typedef __attribute__((ext_vector_type(4)))  unsigned int u32x4;
typedef __attribute__((ext_vector_type(8)))  int    i32x8;
typedef __attribute__((ext_vector_type(4)))  int    i32x4;

#define L_STEPS 1024
#define BATCH   32
#define DIM     1024
#define NGATE   3
#define KSLICES 4
#define KSTEPS  (DIM / 32 / KSLICES)   // 8 K-steps of 32 per wave
#define WSLAB   (NGATE * 16 * DIM)     // 49152 bf16 = 96KB weight slab / block

// ---------------------------------------------------------------------------
// Prep kernel 1: pack weight [D_in, D_out, 3] f32 -> bf16 as wb[ot][g][n][k].
// ---------------------------------------------------------------------------
__global__ __launch_bounds__(256) void pack_weight_kernel(
    const float* __restrict__ w, __bf16* __restrict__ out) {
  int idx = blockIdx.x * 256 + threadIdx.x;       // < 3*1024*1024
  int k  = idx & (DIM - 1);
  int r  = idx >> 10;                             // (ot*3 + g)*16 + n
  int n  = r & 15;
  int gg = r >> 4;
  int g  = gg % 3;
  int ot = gg / 3;
  out[idx] = (__bf16)w[(size_t)k * (NGATE * DIM) + (ot * 16 + n) * NGATE + g];
}

// ---------------------------------------------------------------------------
// Prep kernel 2: x f32 -> bf16 so the GEMM hot loop is pure b128 loads.
// ---------------------------------------------------------------------------
__global__ __launch_bounds__(256) void pack_x_kernel(
    const float* __restrict__ in, __bf16* __restrict__ out) {
  size_t base = ((size_t)blockIdx.x * 256 + threadIdx.x) * 8;
  float4 q0 = *(const float4*)(in + base);
  float4 q1 = *(const float4*)(in + base + 4);
  __bf16 v[8] = {(__bf16)q0.x, (__bf16)q0.y, (__bf16)q0.z, (__bf16)q0.w,
                 (__bf16)q1.x, (__bf16)q1.y, (__bf16)q1.z, (__bf16)q1.w};
  *(uint4*)(out + base) = *(uint4*)v;
}

// ---------------------------------------------------------------------------
// Fused SRU kernel. blockIdx.x: bit0 = batch tile, bits[6:1] = output tile.
// blockDim = 128 (4 waves = 4 K-slices).
// ---------------------------------------------------------------------------
__global__ __launch_bounds__(128) void sru_fused_kernel(
    const float*  __restrict__ x,      // [L,B,D] f32 (highway path)
    const float*  __restrict__ c0,     // [B,D]
    const float*  __restrict__ bias,   // [2D]
    const __bf16* __restrict__ xb,     // [L,B,D] bf16
    const __bf16* __restrict__ wb,     // packed weights
    float* __restrict__ h_out,         // [L,B,D]
    float* __restrict__ c_out)         // [L,B,D]
{
  // wlds declared first -> LDS offset 0 (used as D#.lds_addr for the TDM).
  __shared__ __bf16 wlds[WSLAB];                       // 96 KB
  __shared__ __attribute__((aligned(32)))
      float redbuf[2][KSLICES - 1][NGATE][32][8];      // 18 KB, double-buffered

  const int tid   = threadIdx.x;
  const int lane  = tid & 31;
  const int ks    = tid >> 5;                   // K-slice 0..3 (wave id)
  const int btile = blockIdx.x & 1;
  const int ot    = blockIdx.x >> 1;            // output tile 0..63
  const int b0    = btile * 16;
  const int obase = ot * 16;
  const int n     = lane & 15;                  // N column within tile
  const int hi    = lane >> 4;
  const int mrow0 = hi << 3;                    // C/D: VGPR j <-> row j+mrow0

  // ---- Stage the 96KB weight slab into LDS once, via the Tensor Data Mover.
  const unsigned long long gaddr =
      (unsigned long long)(const void*)(wb + (size_t)ot * WSLAB);
#if __has_builtin(__builtin_amdgcn_tensor_load_to_lds)
  if (ks == 0) {   // TDM ignores EXEC; issue from one wave per block
    u32x4 g0;   // D# group 0: count=1 | lds_addr | global_addr | type=2
    g0[0] = 1u;                                  // count = 1 valid descriptor
    g0[1] = 0u;                                  // lds_addr = 0 (wlds base)
    g0[2] = (unsigned)(gaddr & 0xffffffffull);   // global_addr[31:0]
    g0[3] = (unsigned)((gaddr >> 32) & 0x1ffffffull) | (2u << 30); // type=2
    i32x8 g1;   // D# group 1
    g1[0] = 1 << 16;                 // data_size = 1 (2 bytes), no multicast
    g1[1] = (int)(0xC000u << 16);    // tensor_dim0 = 49152 (bits 63:48)
    g1[2] = 1 << 16;                 // tensor_dim0 hi=0; tensor_dim1 = 1
    g1[3] = (int)(0xC000u << 16);    // tile_dim0 = 49152 (bits 127:112)
    g1[4] = 1;                       // tile_dim1 = 1 row
    g1[5] = (int)0xC000u;            // tensor_dim0_stride = 49152
    g1[6] = 0;
    g1[7] = 0;
    i32x4 z4 = {0, 0, 0, 0};
#if defined(__clang_major__) && (__clang_major__ >= 23)
    i32x8 z8 = {0, 0, 0, 0, 0, 0, 0, 0};
    __builtin_amdgcn_tensor_load_to_lds(g0, g1, z4, z4, z8, 0);
#else
    __builtin_amdgcn_tensor_load_to_lds(g0, g1, z4, z4, 0);
#endif
    __builtin_amdgcn_s_wait_tensorcnt(0);
  }
#else
  {  // Fallback: cooperative plain copy.
    const uint4* src = (const uint4*)(wb + (size_t)ot * WSLAB);
    uint4* dst = (uint4*)wlds;
    for (int i = tid; i < (WSLAB * 2 / 16); i += 128) dst[i] = src[i];
  }
#endif
  // The TDM builtin carries no LDS pointer operand, so without this opaque
  // may-write-memory barrier the optimizer proves wlds is never stored and
  // folds every B-fragment load to undef (deleting the gate WMMA chains).
  asm volatile("" : : "r"(wlds) : "memory");
  __syncthreads();

  // Per-lane gate biases (depend only on the N column)
  const float bias_f = bias[obase + n];
  const float bias_r = bias[DIM + obase + n];

  // Recurrent state (consumer wave only)
  v8f c = {};
  if (ks == 0) {
    #pragma unroll
    for (int j = 0; j < 8; ++j)
      c[j] = c0[(size_t)(b0 + mrow0 + j) * DIM + obase + n];
  }

  // Fragment addressing (this code's contiguous K order; A and B match):
  //  A: lane holds row M=b0+(lane&15); elems 0..15 = K 16*half+0..15.
  //     B: lane = N col, same contiguous K order per half.
  const int arow   = b0 + n;
  const int aoff   = hi << 4;
  const int boff   = hi << 4;
  const int kbase0 = ks * (DIM / KSLICES);      // this wave's K origin

  for (int l = 0; l < L_STEPS; ++l) {
    const __bf16* xrow = xb + ((size_t)l * BATCH + arow) * DIM + kbase0;
    if (l + 1 < L_STEPS)  // prefetch next timestep's A rows
      __builtin_prefetch(xrow + (size_t)BATCH * DIM, 0, 1);

    v8f a0 = {}, a1 = {}, a2 = {};              // u0 / forget / reset partials

    #pragma unroll
    for (int kk = 0; kk < KSTEPS; ++kk) {
      const int kb = kk * 32;
      v16bf A;
      ((uint4*)&A)[0] = *(const uint4*)(xrow + kb + aoff);
      ((uint4*)&A)[1] = *(const uint4*)(xrow + kb + 8 + aoff);
      v16bf B0, B1, B2;
      const __bf16* p0 = wlds + (size_t)(0 * 16 + n) * DIM + kbase0 + kb + boff;
      const __bf16* p1 = wlds + (size_t)(1 * 16 + n) * DIM + kbase0 + kb + boff;
      const __bf16* p2 = wlds + (size_t)(2 * 16 + n) * DIM + kbase0 + kb + boff;
      ((uint4*)&B0)[0] = *(const uint4*)(p0);
      ((uint4*)&B0)[1] = *(const uint4*)(p0 + 8);
      ((uint4*)&B1)[0] = *(const uint4*)(p1);
      ((uint4*)&B1)[1] = *(const uint4*)(p1 + 8);
      ((uint4*)&B2)[0] = *(const uint4*)(p2);
      ((uint4*)&B2)[1] = *(const uint4*)(p2 + 8);
      a0 = __builtin_amdgcn_wmma_f32_16x16x32_bf16(false, A, false, B0,
                                                   (short)0, a0, false, false);
      a1 = __builtin_amdgcn_wmma_f32_16x16x32_bf16(false, A, false, B1,
                                                   (short)0, a1, false, false);
      a2 = __builtin_amdgcn_wmma_f32_16x16x32_bf16(false, A, false, B2,
                                                   (short)0, a2, false, false);
    }

    const int par = l & 1;
    if (ks != 0) {  // producers: publish partial sums for this parity
      *(v8f*)&redbuf[par][ks - 1][0][lane][0] = a0;
      *(v8f*)&redbuf[par][ks - 1][1][lane][0] = a1;
      *(v8f*)&redbuf[par][ks - 1][2][lane][0] = a2;
    }
    __syncthreads();  // one barrier per step; double buffer makes this safe

    if (ks == 0) {    // consumer: reduce, scan, highway, store
      #pragma unroll
      for (int s = 0; s < KSLICES - 1; ++s) {
        a0 += *(const v8f*)&redbuf[par][s][0][lane][0];
        a1 += *(const v8f*)&redbuf[par][s][1][lane][0];
        a2 += *(const v8f*)&redbuf[par][s][2][lane][0];
      }
      #pragma unroll
      for (int j = 0; j < 8; ++j) {
        const size_t off =
            ((size_t)l * BATCH + b0 + mrow0 + j) * DIM + obase + n;
        float u0 = a0[j];
        float f  = 1.0f / (1.0f + __expf(-(a1[j] + bias_f)));
        float r  = 1.0f / (1.0f + __expf(-(a2[j] + bias_r)));
        float cn = f * c[j] + (1.0f - f) * u0;
        c[j] = cn;
        float e  = __expf(-2.0f * cn);
        float g  = (1.0f - e) / (1.0f + e);      // tanh(cn)
        float xv = x[off];
        h_out[off] = r * g + (1.0f - r) * xv;
        c_out[off] = cn;
      }
    }
  }
}

// ---------------------------------------------------------------------------
extern "C" void kernel_launch(void* const* d_in, const int* in_sizes, int n_in,
                              void* d_out, int out_size, void* d_ws, size_t ws_size,
                              hipStream_t stream) {
  const float* x    = (const float*)d_in[0];   // [L,B,D]
  const float* c0   = (const float*)d_in[1];   // [B,D]
  const float* w    = (const float*)d_in[2];   // [D,D,3]
  const float* bias = (const float*)d_in[3];   // [2D]

  float* h_out = (float*)d_out;
  float* c_out = h_out + (size_t)L_STEPS * BATCH * DIM;

  // Workspace: packed bf16 weights (6 MB) then packed bf16 x (64 MB).
  __bf16* wb = (__bf16*)d_ws;
  __bf16* xb = (__bf16*)((char*)d_ws + 8ull * 1024 * 1024);

  pack_weight_kernel<<<(NGATE * DIM * DIM) / 256, 256, 0, stream>>>(w, wb);
  pack_x_kernel<<<((size_t)L_STEPS * BATCH * DIM / 8) / 256, 256, 0, stream>>>(x, xb);
  // 128 blocks = 2 batch-tiles x 64 output-tiles, 4 K-slice waves each.
  sru_fused_kernel<<<2 * (DIM / 16), 128, 0, stream>>>(x, c0, bias, xb, wb,
                                                       h_out, c_out);
}
